// P2M_83837761618528
// MI455X (gfx1250) — compile-verified, matrix operand order checked by reading, not run on hardware
//
#include <hip/hip_runtime.h>
#include <hip/hip_bf16.h>

// ---------------------------------------------------------------------------
// Types for CDNA5 WMMA (wave32): v_wmma_f32_16x16x32_bf16
// ---------------------------------------------------------------------------
typedef __bf16 bf16_t;
typedef bf16_t v16bf __attribute__((ext_vector_type(16)));
typedef float  v8f   __attribute__((ext_vector_type(8)));

__device__ __forceinline__ unsigned short f2bf_raw(float f) {
    unsigned u = __builtin_bit_cast(unsigned, f);
    unsigned r = u + 0x7FFFu + ((u >> 16) & 1u);   // round-to-nearest-even
    return (unsigned short)(r >> 16);
}
__device__ __forceinline__ unsigned pack2bf(float lo, float hi) {
    return (unsigned)f2bf_raw(lo) | ((unsigned)f2bf_raw(hi) << 16);
}

// ---------------------------------------------------------------------------
// Fused gconv GEMM: OUT = act( X @ W0 + AGG @ W1 + bias ) [+ resid]
//   X, AGG : M x K   (fp32, row-major)
//   W0, W1 : K x 128 (fp32, row-major)
//   OUT    : M x 128
// Block = 256 threads (8 waves). Tile: 128 rows x 128 cols.
// Wave w owns cols [16w,16w+16) and 8 row-tiles (B fragment reused 8x).
// LDS holds bf16 packed as u32 pairs; A row-major, B column-major so both
// fragment reads are two contiguous 8-element runs (vectorizable ds loads).
// ---------------------------------------------------------------------------
#define BM 128
#define BN 128
#define BK 32
#define LDA32 17          // u32 stride per A row (16 data + 1 pad)
#define LDB32 17          // u32 stride per B column
#define LDA_BF (LDA32*2)  // bf16 stride
#define LDB_BF (LDB32*2)

__global__ __launch_bounds__(256) void gconv_gemm_kernel(
    const float* __restrict__ X, const float* __restrict__ AGG,
    const float* __restrict__ W0, const float* __restrict__ W1,
    const float* __restrict__ bias, const float* __restrict__ resid,
    float* __restrict__ OUT, int M, int K, int do_relu)
{
    __shared__ unsigned AsU[BM * LDA32];   // 128 rows x 32 bf16
    __shared__ unsigned BtU[BN * LDB32];   // 128 cols x 32 bf16 (transposed)
    const bf16_t* As = (const bf16_t*)AsU;
    const bf16_t* Bt = (const bf16_t*)BtU;

    const int tid  = threadIdx.x;
    const int wave = tid >> 5;
    const int lane = tid & 31;
    const int lidx = lane & 15;   // row (A) / col (B,C) within 16-tile
    const int kh   = lane >> 4;   // which k-half this lane group holds
    const int row0 = blockIdx.x * BM;

    // A-staging map: each thread stages one half-row (16 bf16 = 8 u32)
    const int a_row  = tid >> 1;
    const int a_half = tid & 1;
    // B-staging map: each thread stages 16 k's of one column (8 u32)
    const int b_col = tid & 127;
    const int b_kg  = tid >> 7;

    v8f acc[8] = {};

    for (int p = 0; p < 2; ++p) {
        const float* A = p ? AGG : X;
        const float* W = p ? W1  : W0;
        for (int k0 = 0; k0 < K; k0 += BK) {
            // ---- stage A: row a_row, k = k0 + a_half*16 .. +15 ----
            {
                const int gr = row0 + a_row;
                const int kb = k0 + a_half * 16;
                unsigned* dstp = &AsU[a_row * LDA32 + a_half * 8];
                if (gr < M && kb + 16 <= K) {
                    const float* srcp = A + (long)gr * K + kb;
                    #pragma unroll
                    for (int j = 0; j < 8; ++j)
                        dstp[j] = pack2bf(srcp[2 * j], srcp[2 * j + 1]);
                } else {
                    const float* srcp = A + (long)gr * K + kb;
                    #pragma unroll
                    for (int j = 0; j < 8; ++j) {
                        float a0 = (gr < M && kb + 2 * j     < K) ? srcp[2 * j]     : 0.0f;
                        float a1 = (gr < M && kb + 2 * j + 1 < K) ? srcp[2 * j + 1] : 0.0f;
                        dstp[j] = pack2bf(a0, a1);
                    }
                }
            }
            // ---- stage B transposed: col b_col, k = k0 + b_kg*16 .. +15 ----
            {
                const int kb = k0 + b_kg * 16;
                unsigned* dstp = &BtU[b_col * LDB32 + b_kg * 8];
                if (kb + 16 <= K) {
                    const float* srcp = W + (long)kb * BN + b_col;
                    #pragma unroll
                    for (int j = 0; j < 8; ++j)
                        dstp[j] = pack2bf(srcp[(2 * j) * BN], srcp[(2 * j + 1) * BN]);
                } else {
                    const float* srcp = W + (long)kb * BN + b_col;
                    #pragma unroll
                    for (int j = 0; j < 8; ++j) {
                        float b0 = (kb + 2 * j     < K) ? srcp[(2 * j) * BN]     : 0.0f;
                        float b1 = (kb + 2 * j + 1 < K) ? srcp[(2 * j + 1) * BN] : 0.0f;
                        dstp[j] = pack2bf(b0, b1);
                    }
                }
            }
            // prefetch next K-slab of A (global_prefetch_b8)
            if (k0 + BK < K) {
                int pr = row0 + (tid >> 1);
                int pc = k0 + BK + (tid & 1) * 16;
                if (pr < M && pc < K)
                    __builtin_prefetch(&A[(long)pr * K + pc], 0, 1);
            }
            __syncthreads();

            // ---- B fragment: two contiguous 8-elem runs per lane ----
            v16bf bfrag;
            {
                const bf16_t* bp = Bt + (wave * 16 + lidx) * LDB_BF + kh * 8;
                #pragma unroll
                for (int e = 0; e < 8; ++e) bfrag[e] = bp[e];
                #pragma unroll
                for (int e = 0; e < 8; ++e) bfrag[8 + e] = bp[16 + e];
            }
            // ---- 8 row-tiles: A fragment + WMMA accumulate ----
            #pragma unroll
            for (int r = 0; r < 8; ++r) {
                v16bf afrag;
                const bf16_t* ap = As + (r * 16 + lidx) * LDA_BF + kh * 8;
                #pragma unroll
                for (int e = 0; e < 8; ++e) afrag[e] = ap[e];
                #pragma unroll
                for (int e = 0; e < 8; ++e) afrag[8 + e] = ap[16 + e];
                acc[r] = __builtin_amdgcn_wmma_f32_16x16x32_bf16(
                    /*neg_a=*/false, afrag, /*neg_b=*/false, bfrag,
                    /*c_mod=*/(short)0, acc[r],
                    /*reuse_a=*/false, /*reuse_b=*/false);
            }
            __syncthreads();
        }
    }

    // ---- epilogue: C/D layout lane = N, vgpr v -> M = v + 8*(lane>=16) ----
    const int col = wave * 16 + lidx;
    const float bv = bias[col];
    #pragma unroll
    for (int r = 0; r < 8; ++r) {
        #pragma unroll
        for (int v = 0; v < 8; ++v) {
            int row = row0 + r * 16 + v + kh * 8;
            if (row < M) {
                float val = acc[r][v] + bv;
                if (do_relu) val = fmaxf(val, 0.0f);
                if (resid)   val += resid[(long)row * BN + col];
                OUT[(long)row * BN + col] = val;
            }
        }
    }
}

// ---------------------------------------------------------------------------
// Elementwise / graph kernels
// ---------------------------------------------------------------------------
__global__ void zero_kernel(float* __restrict__ p, long n) {
    for (long i = blockIdx.x * (long)blockDim.x + threadIdx.x; i < n;
         i += (long)gridDim.x * blockDim.x)
        p[i] = 0.0f;
}

// agg[src[e]] += x[dst[e]]  (segment_sum over directed edges), 2D grid
__global__ void edge_agg_kernel(const float* __restrict__ x, float* __restrict__ agg,
                                const int* __restrict__ src, const int* __restrict__ dst,
                                int E, int Kw)
{
    int d = blockIdx.x * blockDim.x + threadIdx.x;
    if (d >= Kw) return;
    for (int e = blockIdx.y; e < E; e += gridDim.y) {
        int s = src[e], t = dst[e];
        atomicAdd(&agg[(long)s * Kw + d], x[(long)t * Kw + d]);
    }
}

// pix = project(verts): P rows [fx,0,cx,.8cx],[0,fy,cy,.8cy],[0,0,1,.8]
__global__ void project_kernel(const float* __restrict__ verts,
                               const float* __restrict__ cc, const float* __restrict__ cf,
                               float* __restrict__ pix, int N)
{
    int i = blockIdx.x * blockDim.x + threadIdx.x;
    if (i >= N) return;
    float X = verts[i * 3 + 0], Y = verts[i * 3 + 1], Z = verts[i * 3 + 2];
    float fx = cf[0], fy = cf[1], cx = cc[0], cy = cc[1];
    float w = Z + 0.8f;
    pix[i * 2 + 0] = (fx * X + cx * Z + 0.8f * cx) / w;
    pix[i * 2 + 1] = (fy * Y + cy * Z + 0.8f * cy) / w;
}

// feats[v, 0:1280) = bilerp(f3|f4|f5); feats[v, 1280:FW) = sfeat
__global__ void feats_kernel(const float* __restrict__ pix,
                             const float* __restrict__ f3, const float* __restrict__ f4,
                             const float* __restrict__ f5,
                             const float* __restrict__ sfeat, int sfW,
                             float* __restrict__ feats, int FW,
                             const int* __restrict__ image_size_p, long N)
{
    const int IS = *image_size_p;
    long total = N * (long)FW;
    for (long i = blockIdx.x * (long)blockDim.x + threadIdx.x; i < total;
         i += (long)gridDim.x * blockDim.x) {
        long v = i / FW;
        int  c = (int)(i - v * FW);
        float out;
        if (c < 1280) {
            const float* f; int dim, cc2;
            if      (c < 256) { f = f3; dim = 56; cc2 = c; }
            else if (c < 768) { f = f4; dim = 28; cc2 = c - 256; }
            else              { f = f5; dim = 14; cc2 = c - 768; }
            float px = pix[v * 2 + 0], py = pix[v * 2 + 1];
            float x = px * dim / IS, y = py * dim / IS;
            float x1 = floorf(x), y1 = floorf(y);
            float x2 = x1 + 1.0f, y2 = y1 + 1.0f;
            int xi1 = min(max((int)x1, 0), dim - 1);
            int xi2 = min(max((int)x2, 0), dim - 1);
            int yi1 = min(max((int)y1, 0), dim - 1);
            int yi2 = min(max((int)y2, 0), dim - 1);
            const float* fb = f + (long)cc2 * dim * dim;
            float f11 = fb[xi1 * dim + yi1], f12 = fb[xi1 * dim + yi2];
            float f21 = fb[xi2 * dim + yi1], f22 = fb[xi2 * dim + yi2];
            out = (x2 - x) * (y2 - y) * f11 + (x2 - x) * (y - y1) * f12 +
                  (x - x1) * (y2 - y) * f21 + (x - x1) * (y - y1) * f22;
        } else {
            out = sfeat[v * sfW + (c - 1280)];
        }
        feats[i] = out;
    }
}

// coords = h @ Wc0 + aggh @ Wc1 + bc   (128 -> 3, tiny)
__global__ void coords_kernel(const float* __restrict__ h, const float* __restrict__ aggh,
                              const float* __restrict__ Wc0, const float* __restrict__ Wc1,
                              const float* __restrict__ bc, float* __restrict__ out, long N)
{
    long total = N * 3;
    for (long i = blockIdx.x * (long)blockDim.x + threadIdx.x; i < total;
         i += (long)gridDim.x * blockDim.x) {
        long v = i / 3;
        int  j = (int)(i - v * 3);
        float s = bc[j];
        const float* hr = h + v * 128;
        const float* ar = aggh + v * 128;
        #pragma unroll 4
        for (int k = 0; k < 128; ++k)
            s += hr[k] * Wc0[k * 3 + j] + ar[k] * Wc1[k * 3 + j];
        out[i] = s;
    }
}

// unpool: rows [0,N) copied; rows [N,N+Nm) = midpoint avg of mid pairs
__global__ void unpool_kernel(const float* __restrict__ coords, const float* __restrict__ feat,
                              const int* __restrict__ mid,
                              float* __restrict__ verts_out, float* __restrict__ feat_out,
                              int N, int Nm)
{
    long total = (long)(N + Nm) * 131;
    for (long i = blockIdx.x * (long)blockDim.x + threadIdx.x; i < total;
         i += (long)gridDim.x * blockDim.x) {
        long row = i / 131;
        int  c = (int)(i - row * 131);
        float val;
        if (row < N) {
            val = (c < 3) ? coords[row * 3 + c] : feat[row * 128 + (c - 3)];
        } else {
            int m = (int)(row - N);
            int a = mid[m * 2 + 0], b = mid[m * 2 + 1];
            float va = (c < 3) ? coords[a * 3 + c] : feat[(long)a * 128 + (c - 3)];
            float vb = (c < 3) ? coords[b * 3 + c] : feat[(long)b * 128 + (c - 3)];
            val = 0.5f * (va + vb);
        }
        if (c < 3) verts_out[row * 3 + c] = val;
        else       feat_out[row * 128 + (c - 3)] = val;
    }
}

// ---------------------------------------------------------------------------
// Host-side stage orchestration (all launches on `stream`, no sync)
// ---------------------------------------------------------------------------
static void launch_edge_agg(hipStream_t stream, const float* x, float* agg,
                            const int* src, const int* dst, long E, int Kw)
{
    dim3 grid((Kw + 255) / 256, (unsigned)((E < 32768) ? E : 32768));
    edge_agg_kernel<<<grid, 256, 0, stream>>>(x, agg, src, dst, (int)E, Kw);
}

static void run_stage(hipStream_t stream, int N, long E, const int* edges,
                      const float* verts_in, const float* sfeat_in, int sfW, int FW,
                      const float* Win0, const float* Win1, const float* bin,
                      const float* W0s, const float* W1s, const float* bs,
                      const float* Wc0, const float* Wc1, const float* bc,
                      const float* f3, const float* f4, const float* f5,
                      const float* cc, const float* cf, const int* isz,
                      float* feats, float* agg, float* hA, float* hB, float* pix,
                      float* coords_out, float** h_out)
{
    const int* src = edges;
    const int* dst = edges + E;

    project_kernel<<<(N + 255) / 256, 256, 0, stream>>>(verts_in, cc, cf, pix, N);
    feats_kernel<<<2048, 256, 0, stream>>>(pix, f3, f4, f5, sfeat_in, sfW, feats, FW, isz, N);

    zero_kernel<<<2048, 256, 0, stream>>>(agg, (long)N * FW);
    launch_edge_agg(stream, feats, agg, src, dst, E, FW);
    gconv_gemm_kernel<<<(N + BM - 1) / BM, 256, 0, stream>>>(
        feats, agg, Win0, Win1, bin, nullptr, hA, N, FW, 1);

    float* cur = hA; float* alt = hB;
    for (int l = 0; l < 12; ++l) {
        zero_kernel<<<1024, 256, 0, stream>>>(agg, (long)N * 128);
        launch_edge_agg(stream, cur, agg, src, dst, E, 128);
        gconv_gemm_kernel<<<(N + BM - 1) / BM, 256, 0, stream>>>(
            cur, agg, W0s + (long)l * 128 * 128, W1s + (long)l * 128 * 128,
            bs + (long)l * 128, cur, alt, N, 128, 1);
        float* t = cur; cur = alt; alt = t;
    }

    zero_kernel<<<1024, 256, 0, stream>>>(agg, (long)N * 128);
    launch_edge_agg(stream, cur, agg, src, dst, E, 128);
    coords_kernel<<<1024, 256, 0, stream>>>(cur, agg, Wc0, Wc1, bc, coords_out, (long)N);
    *h_out = cur;
}

extern "C" void kernel_launch(void* const* d_in, const int* in_sizes, int n_in,
                              void* d_out, int out_size, void* d_ws, size_t ws_size,
                              hipStream_t stream)
{
    // ---- inputs (setup_inputs order) ----
    const float* f3 = (const float*)d_in[0];
    const float* f4 = (const float*)d_in[1];
    const float* f5 = (const float*)d_in[2];
    const float* verts0 = (const float*)d_in[3];
    const float* cam_c  = (const float*)d_in[4];
    const float* cam_f  = (const float*)d_in[5];
    const float* g[3][9];
    for (int s = 0; s < 3; ++s)
        for (int k = 0; k < 9; ++k)
            g[s][k] = (const float*)d_in[6 + s * 9 + k];
    const int* edges1 = (const int*)d_in[33];
    const int* edges2 = (const int*)d_in[34];
    const int* edges3 = (const int*)d_in[35];
    const int* mid0   = (const int*)d_in[36];
    const int* mid1   = (const int*)d_in[37];
    const int* isz    = (const int*)d_in[38];

    const int  N1  = in_sizes[3] / 3;
    const long E1  = in_sizes[33] / 2;
    const long E2  = in_sizes[34] / 2;
    const long E3  = in_sizes[35] / 2;
    const int  Nm0 = in_sizes[36] / 2;
    const int  Nm1 = in_sizes[37] / 2;
    const int  N2  = N1 + Nm0;
    const int  N3  = N2 + Nm1;

    // ---- workspace layout ----
    size_t off = 0;
    auto wsalloc = [&](size_t bytes) {
        void* p = (char*)d_ws + off;
        off += (bytes + 255) & ~(size_t)255;
        return p;
    };
    float* feats  = (float*)wsalloc((size_t)N3 * 1408 * 4);
    float* agg    = (float*)wsalloc((size_t)N3 * 1408 * 4);
    float* hA     = (float*)wsalloc((size_t)N3 * 128 * 4);
    float* hB     = (float*)wsalloc((size_t)N3 * 128 * 4);
    float* sfeat  = (float*)wsalloc((size_t)N3 * 128 * 4);
    float* verts  = (float*)wsalloc((size_t)N3 * 3 * 4);
    float* coords = (float*)wsalloc((size_t)N3 * 3 * 4);
    float* pix    = (float*)wsalloc((size_t)N3 * 2 * 4);

    float* hout = nullptr;

    // ---- stage 1: N1 verts, shape feat = verts0 (3), FW=1283 ----
    run_stage(stream, N1, E1, edges1, verts0, verts0, 3, 1283,
              g[0][0], g[0][1], g[0][2], g[0][3], g[0][4], g[0][5],
              g[0][6], g[0][7], g[0][8],
              f3, f4, f5, cam_c, cam_f, isz,
              feats, agg, hA, hB, pix, coords, &hout);
    unpool_kernel<<<2048, 256, 0, stream>>>(coords, hout, mid0, verts, sfeat, N1, Nm0);

    // ---- stage 2: N2 verts, shape feat = unpooled h (128), FW=1408 ----
    run_stage(stream, N2, E2, edges2, verts, sfeat, 128, 1408,
              g[1][0], g[1][1], g[1][2], g[1][3], g[1][4], g[1][5],
              g[1][6], g[1][7], g[1][8],
              f3, f4, f5, cam_c, cam_f, isz,
              feats, agg, hA, hB, pix, coords, &hout);
    unpool_kernel<<<2048, 256, 0, stream>>>(coords, hout, mid1, verts, sfeat, N2, Nm1);

    // ---- stage 3: N3 verts, final coords straight to d_out ----
    run_stage(stream, N3, E3, edges3, verts, sfeat, 128, 1408,
              g[2][0], g[2][1], g[2][2], g[2][3], g[2][4], g[2][5],
              g[2][6], g[2][7], g[2][8],
              f3, f4, f5, cam_c, cam_f, isz,
              feats, agg, hA, hB, pix, (float*)d_out, &hout);
    (void)n_in; (void)out_size; (void)ws_size;
}
